// Projector_51788715655680
// MI455X (gfx1250) — matrix operand order repaired and checked by reading, not compiled, not run on hardware
//
#include <hip/hip_runtime.h>
#include <math.h>

typedef __attribute__((ext_vector_type(16))) _Float16 v16h;
typedef __attribute__((ext_vector_type(8)))  _Float16 v8h;
typedef __attribute__((ext_vector_type(8)))  float    v8f;

#define EPSV 1e-5f
#define S_PIX 16384   // H*W = 128*128
#define L_TOK 256
#define EPAD  272     // 256 + 16 halfs: LDS row stride, avoids bank-0 pileup

// ---------------------------------------------------------------------------
// Kernel 1: pk/pv = BN(1x1 conv1d(tokens)).  Tiny (67 MFLOP) -> simple FMA.
// Writes pk as f32 (consumed by kernel 2) and pv as f16 [n][c][l] (V operand).
// ---------------------------------------------------------------------------
__global__ void pkv_kernel(const float* __restrict__ tokens,
                           const float* __restrict__ w_v, const float* __restrict__ b_v,
                           const float* __restrict__ g_v, const float* __restrict__ be_v,
                           const float* __restrict__ m_v, const float* __restrict__ v_v,
                           const float* __restrict__ w_k, const float* __restrict__ b_k,
                           const float* __restrict__ g_k, const float* __restrict__ be_k,
                           const float* __restrict__ m_k, const float* __restrict__ v_k,
                           float* __restrict__ pk, _Float16* __restrict__ vh)
{
  __shared__ float wk_s[256], wv_s[256];
  int c = blockIdx.x & 255, n = blockIdx.x >> 8;
  int l = threadIdx.x;
  wk_s[l] = w_k[c * 256 + l];
  wv_s[l] = w_v[c * 256 + l];
  __syncthreads();
  const float* tok = tokens + (size_t)n * 256 * 256 + l;   // column l, coalesced over l
  float ak = 0.f, av = 0.f;
  for (int t = 0; t < 256; ++t) {
    float x = tok[(size_t)t * 256];
    ak = fmaf(wk_s[t], x, ak);
    av = fmaf(wv_s[t], x, av);
  }
  float sk = g_k[c] / sqrtf(v_k[c] + EPSV);
  float sv = g_v[c] / sqrtf(v_v[c] + EPSV);
  pk[(size_t)(n * 256 + c) * 256 + l] = (ak + b_k[c] - m_k[c]) * sk + be_k[c];
  vh[(size_t)(n * 256 + c) * 256 + l] = (_Float16)((av + b_v[c] - m_v[c]) * sv + be_v[c]);
}

// ---------------------------------------------------------------------------
// Kernel 2: fold BN_q + grouped conv w_q into k, producing a padded K=32 block
// per token l (f16, contiguous):
//   kq32[n,h,l,ci]   = 0.25 * sum_d sq[d] * w_q[h,d,ci] * pk[n,h*16+d,l]  (ci<16)
//   kq32[n,h,l,16]   = 0.25 * sum_d cq[d] * pk[n,h*16+d,l]   (bias row; A has 1.0)
//   kq32[n,h,l,17..31] = 0
// This lets the score WMMA run with a full-width B fragment (two 16B loads per
// lane), inline-0 C, and zero masking/broadcast VALU.
// ---------------------------------------------------------------------------
__global__ void kq_kernel(const float* __restrict__ pk,
                          const float* __restrict__ w_q, const float* __restrict__ b_q,
                          const float* __restrict__ g_q, const float* __restrict__ be_q,
                          const float* __restrict__ m_q, const float* __restrict__ v_q,
                          _Float16* __restrict__ kq32)
{
  __shared__ float wq_s[16][16], sq_s[16], cq_s[16];
  int h = blockIdx.x & 15, n = blockIdx.x >> 4;
  int tid = threadIdx.x;
  wq_s[tid >> 4][tid & 15] = w_q[(size_t)(h * 16 + (tid >> 4)) * 16 + (tid & 15)];
  if (tid < 16) {
    int cc = h * 16 + tid;
    float s = g_q[cc] / sqrtf(v_q[cc] + EPSV);
    sq_s[tid] = 0.25f * s;                                    // fold 1/sqrt(dq)=0.25
    cq_s[tid] = 0.25f * ((b_q[cc] - m_q[cc]) * s + be_q[cc]);
  }
  __syncthreads();
  int l = tid;
  float pkd[16];
#pragma unroll
  for (int d = 0; d < 16; ++d)
    pkd[d] = pk[(size_t)((n * 16 + h) * 16 + d) * 256 + l];
  _Float16* o = kq32 + ((size_t)(n * 16 + h) * 256 + l) * 32;
#pragma unroll
  for (int ci = 0; ci < 16; ++ci) {
    float acc = 0.f;
#pragma unroll
    for (int d = 0; d < 16; ++d) acc = fmaf(sq_s[d] * wq_s[d][ci], pkd[d], acc);
    o[ci] = (_Float16)acc;
  }
  float bias = 0.f;
#pragma unroll
  for (int d = 0; d < 16; ++d) bias = fmaf(cq_s[d], pkd[d], bias);
  o[16] = (_Float16)bias;
#pragma unroll
  for (int e = 17; e < 32; ++e) o[e] = (_Float16)0.f;
}

// ---------------------------------------------------------------------------
// Kernel 3: fused attention. One wave handles one (n,h,16-pixel tile):
//   scores (16 WMMAs, full K=32: 16 ci + bias-via-1 + zero pad, inline-0 C) ->
//   rowwise softmax (shfl_xor inside 16-lane halves, f32 exp)               ->
//   E -> LDS [s][l] f16 (scalar b16 stores, padded rows)                    ->
//   P^T = V * E^T (8 WMMAs; B-frags are two v8h LDS loads each)   ->
//   coalesced residual-add stores (feature tile re-read hits WGP$).
// ---------------------------------------------------------------------------
__launch_bounds__(128)
__global__ void attn_kernel(const float* __restrict__ feature,
                            const _Float16* __restrict__ vh,
                            const _Float16* __restrict__ kq32,
                            float* __restrict__ out)
{
  __shared__ __align__(64) _Float16 eLds[4][16 * EPAD];  // per-wave E, layout [s][l]
  __shared__ float rLds[4][16];                          // per-wave 1/rowsum by s

  const int S = S_PIX, L = L_TOK;
  int lane = threadIdx.x & 31;
  int wave = threadIdx.x >> 5;
  int m = lane & 15, kh = lane >> 4;
  int h = blockIdx.y, n = blockIdx.z;
  int sBase = blockIdx.x * 64 + wave * 16;

  // ---- A fragment: transposed feature tile, K = ci; k=16 slot carries 1.0 ----
  // ISA 16-bit A 16x32 layout: elem e<8 -> K = e + 8*kh ; e>=8 -> K = 16+(e-8)+8*kh
  const float* fb = feature + ((size_t)(n * 256 + h * 16)) * S + sBase + m;
  v16h a;
#pragma unroll
  for (int e = 0; e < 8; ++e) a[e] = (_Float16)fb[(size_t)(e + 8 * kh) * S];
  a[8] = kh ? (_Float16)0.f : (_Float16)1.0f;   // K=16 -> bias row of kq32
#pragma unroll
  for (int e = 9; e < 16; ++e) a[e] = (_Float16)0.f;

  // ---- scores: 16 tiles cover L=256; B = kq32[l][kh*8 .. +7] and [16+kh*8 .. +7] ----
  const _Float16* kqb = kq32 + (size_t)(n * 16 + h) * L * 32;
  v8f acc[16];
  const v8f zc = {};   // inline 0 accumulator
#pragma unroll
  for (int t = 0; t < 16; ++t) {
    v8h blo = *(const v8h*)(kqb + (size_t)(t * 16 + m) * 32 + kh * 8);
    v8h bhi = *(const v8h*)(kqb + (size_t)(t * 16 + m) * 32 + 16 + kh * 8);
    v16h b;
#pragma unroll
    for (int e = 0; e < 8; ++e) { b[e] = blo[e]; b[e + 8] = bhi[e]; }
    acc[t] = __builtin_amdgcn_wmma_f32_16x16x32_f16(false, a, false, b, (short)0, zc,
                                                    false, false);
  }

  // ---- softmax over L; row s = r + 8*kh lives in one 16-lane half ----
  float rmax[8], rsum[8];
#pragma unroll
  for (int r = 0; r < 8; ++r) rmax[r] = acc[0][r];
#pragma unroll
  for (int t = 1; t < 16; ++t)
#pragma unroll
    for (int r = 0; r < 8; ++r) rmax[r] = fmaxf(rmax[r], acc[t][r]);
#pragma unroll
  for (int r = 0; r < 8; ++r) {
#pragma unroll
    for (int x = 8; x > 0; x >>= 1)
      rmax[r] = fmaxf(rmax[r], __shfl_xor(rmax[r], x, 32));
    rsum[r] = 0.f;
  }
#pragma unroll
  for (int t = 0; t < 16; ++t)
#pragma unroll
    for (int r = 0; r < 8; ++r) {
      float e = __expf(acc[t][r] - rmax[r]);
      acc[t][r] = e;             // keep unnormalized; divide after the V matmul
      rsum[r] += e;
    }
#pragma unroll
  for (int r = 0; r < 8; ++r)
#pragma unroll
    for (int x = 8; x > 0; x >>= 1)
      rsum[r] += __shfl_xor(rsum[r], x, 32);

  // ---- E -> LDS [s][l] (row stride EPAD): scalar f16 stores ----
  _Float16* eL = &eLds[wave][0];
#pragma unroll
  for (int t = 0; t < 16; ++t)
#pragma unroll
    for (int r = 0; r < 8; ++r)
      eL[(r + 8 * kh) * EPAD + t * 16 + m] = (_Float16)acc[t][r];
  if (m == 0) {   // lanes 0 and 16: publish 1/rowsum indexed by s
#pragma unroll
    for (int r = 0; r < 8; ++r) rLds[wave][r + 8 * kh] = 1.0f / rsum[r];
  }
  __syncthreads();

  // ---- P^T[dv,s] = sum_l V[dv,l] * E[s,l] ; K = l in 8 chunks of 32 ----
  const _Float16* vb = vh + ((size_t)(n * 256 + h * 16) + m) * L;   // row dv = m
  v8f p = {};
#pragma unroll
  for (int c = 0; c < 8; ++c) {
    v8h lo = *(const v8h*)(vb + c * 32 + 8 * kh);        // A elems 0-7:  l = c*32+8kh+e
    v8h hi = *(const v8h*)(vb + c * 32 + 16 + 8 * kh);   // A elems 8-15: l = +16
    v16h av;
#pragma unroll
    for (int e = 0; e < 8; ++e) { av[e] = lo[e]; av[e + 8] = hi[e]; }
    // B: lane col = s = m, K elem e<8 -> l = c*32 + 8*kh + e ; e>=8 -> +16
    v8h elo = *(const v8h*)(eL + m * EPAD + c * 32 + 8 * kh);
    v8h ehi = *(const v8h*)(eL + m * EPAD + c * 32 + 16 + 8 * kh);
    v16h bE;
#pragma unroll
    for (int e = 0; e < 8; ++e) { bE[e] = elo[e]; bE[e + 8] = ehi[e]; }
    p = __builtin_amdgcn_wmma_f32_16x16x32_f16(false, av, false, bE, (short)0, p,
                                               false, false);
  }

  // ---- normalize, residual add, coalesced store (D: M=dv=r+8kh, N=s=m) ----
  float rinv = rLds[wave][m];
  const float* fo = feature + ((size_t)(n * 256 + h * 16)) * S + sBase + m;
  float* oo = out + ((size_t)(n * 256 + h * 16)) * S + sBase + m;
#pragma unroll
  for (int r = 0; r < 8; ++r) {
    size_t idx = (size_t)(r + 8 * kh) * S;
    oo[idx] = fo[idx] + p[r] * rinv;
  }
}

// ---------------------------------------------------------------------------
extern "C" void kernel_launch(void* const* d_in, const int* in_sizes, int n_in,
                              void* d_out, int out_size, void* d_ws, size_t ws_size,
                              hipStream_t stream) {
  const float* feature = (const float*)d_in[0];
  const float* tokens  = (const float*)d_in[1];
  const float* w_v  = (const float*)d_in[2];
  const float* b_v  = (const float*)d_in[3];
  const float* g_v  = (const float*)d_in[4];
  const float* be_v = (const float*)d_in[5];
  const float* m_v  = (const float*)d_in[6];
  const float* v_v  = (const float*)d_in[7];
  const float* w_k  = (const float*)d_in[8];
  const float* b_k  = (const float*)d_in[9];
  const float* g_k  = (const float*)d_in[10];
  const float* be_k = (const float*)d_in[11];
  const float* m_k  = (const float*)d_in[12];
  const float* v_k  = (const float*)d_in[13];
  const float* w_q  = (const float*)d_in[14];
  const float* b_q  = (const float*)d_in[15];
  const float* g_q  = (const float*)d_in[16];
  const float* be_q = (const float*)d_in[17];
  const float* m_q  = (const float*)d_in[18];
  const float* v_q  = (const float*)d_in[19];

  // workspace: pk f32 (512KB) | v f16 (256KB) | kq32 f16 (512KB)
  char* ws = (char*)d_ws;
  float*    pk   = (float*)ws;
  _Float16* vhh  = (_Float16*)(ws + 512 * 1024);
  _Float16* kq32 = (_Float16*)(ws + 768 * 1024);

  pkv_kernel<<<dim3(512), dim3(256), 0, stream>>>(tokens,
      w_v, b_v, g_v, be_v, m_v, v_v,
      w_k, b_k, g_k, be_k, m_k, v_k, pk, vhh);
  kq_kernel<<<dim3(32), dim3(256), 0, stream>>>(pk,
      w_q, b_q, g_q, be_q, m_q, v_q, kq32);
  attn_kernel<<<dim3(256, 16, 2), dim3(128), 0, stream>>>(feature, vhh, kq32,
      (float*)d_out);
}
